// LocalMHAAttention2D_59502476919353
// MI455X (gfx1250) — matrix operand order repaired and checked by reading, not compiled, not run on hardware
//
#include <hip/hip_runtime.h>
#include <hip/hip_bf16.h>
#include <math.h>

typedef __bf16 bf16;
typedef __attribute__((ext_vector_type(4)))  bf16  v4bf;
typedef __attribute__((ext_vector_type(8)))  bf16  v8bf;
typedef __attribute__((ext_vector_type(16))) bf16  v16bf;
typedef __attribute__((ext_vector_type(8)))  float v8f;

#define HW   65536   // H*W
#define CCH  256
#define BB   4
#define GG   32
#define WIN  1024    // (H/8)*(W/8)

static __device__ __forceinline__ v8f zero8() {
  v8f z;
#pragma unroll
  for (int i = 0; i < 8; ++i) z[i] = 0.0f;
  return z;
}

// A fragment (16x32 bf16) from row-major [row][k] source, ld elements per row.
// Lane<16: elems 0..7 = K klo..klo+7, elems 8..15 = K 16+klo.. (klo = 8*(lane>=16))
static __device__ __forceinline__ v16bf frag_a(const bf16* p, int ld, int row, int k0, int lane) {
  const int klo = (lane >> 4) << 3;
  const bf16* r = p + (size_t)row * ld + k0 + klo;
  v8bf lo = *(const v8bf*)(r);
  v8bf hi = *(const v8bf*)(r + 16);
  v16bf a;
#pragma unroll
  for (int i = 0; i < 8; ++i) { a[i] = lo[i]; a[8 + i] = hi[i]; }
  return a;
}

// B fragment (32x16 bf16) from K-contiguous [n][k] source.
// Lane<16: elems 0..7 = K kb..kb+7, elems 8..15 = K 16+kb.. (kb = 8*(lane>=16))
static __device__ __forceinline__ v16bf frag_b(const bf16* p, int ld, int col, int k0, int lane) {
  const int kb = (lane >> 4) << 3;
  const bf16* r = p + (size_t)col * ld + k0 + kb;
  v8bf lo = *(const v8bf*)(r);
  v8bf hi = *(const v8bf*)(r + 16);
  v16bf b;
#pragma unroll
  for (int i = 0; i < 8; ++i) { b[i] = lo[i]; b[8 + i] = hi[i]; }
  return b;
}

#define WMMA_BF16(a, b, c) \
  __builtin_amdgcn_wmma_f32_16x16x32_bf16(false, (a), false, (b), (short)0, (c), false, false)

// ---------------- GroupNorm stats: one block per (b, g) ----------------
__global__ __launch_bounds__(256) void k_gn_stats(const float* __restrict__ x,
                                                  float* __restrict__ stats) {
  __shared__ float rs[256], rs2[256];
  const int b = blockIdx.x >> 5, g = blockIdx.x & 31;
  float s = 0.f, s2 = 0.f;
  for (int ci = 0; ci < 8; ++ci) {
    const float* p = x + (((size_t)(b * CCH + g * 8 + ci)) << 16);
    for (int i = threadIdx.x; i < HW; i += 256) { float v = p[i]; s += v; s2 += v * v; }
  }
  rs[threadIdx.x] = s; rs2[threadIdx.x] = s2;
  __syncthreads();
  for (int o = 128; o > 0; o >>= 1) {
    if (threadIdx.x < o) {
      rs[threadIdx.x]  += rs[threadIdx.x + o];
      rs2[threadIdx.x] += rs2[threadIdx.x + o];
    }
    __syncthreads();
  }
  if (threadIdx.x == 0) {
    const float inv_n = 1.0f / (8.0f * (float)HW);
    float mean = rs[0] * inv_n;
    float var  = rs2[0] * inv_n - mean * mean;
    stats[(b * GG + g) * 2 + 0] = mean;
    stats[(b * GG + g) * 2 + 1] = rsqrtf(var + 1e-6f);
  }
}

// ---------------- normalize + affine -> bf16 ----------------
__global__ __launch_bounds__(256) void k_norm(const float* __restrict__ x,
                                              const float* __restrict__ gw,
                                              const float* __restrict__ gb,
                                              const float* __restrict__ stats,
                                              bf16* __restrict__ xn) {
  const size_t i0 = (((size_t)blockIdx.x * 256) + threadIdx.x) * 4;
  const int c = (int)((i0 >> 16) & 255);
  const int b = (int)(i0 >> 24);
  const int g = c >> 3;
  const float mean = stats[(b * GG + g) * 2 + 0];
  const float rstd = stats[(b * GG + g) * 2 + 1];
  const float w = gw[c] * rstd, bia = gb[c] - mean * w;
  const float4 v = *(const float4*)(x + i0);
  v4bf o;
  o[0] = (bf16)(v.x * w + bia);
  o[1] = (bf16)(v.y * w + bia);
  o[2] = (bf16)(v.z * w + bia);
  o[3] = (bf16)(v.w * w + bia);
  *(v4bf*)(xn + i0) = o;
}

// ---------------- weight f32 -> bf16 (q,k,v,o concatenated) ----------------
__global__ __launch_bounds__(256) void k_wconv(const float* __restrict__ wq,
                                               const float* __restrict__ wk,
                                               const float* __restrict__ wv,
                                               const float* __restrict__ wo,
                                               bf16* __restrict__ dst) {
  const int i = blockIdx.x * 256 + threadIdx.x;
  const int mat = i >> 16, off = i & 65535;
  const float* src = (mat == 0) ? wq : (mat == 1) ? wk : (mat == 2) ? wv : wo;
  dst[i] = (bf16)src[off];
}

// ---------------- fused QKV projection + windowed attention ----------------
// Grid: B * WIN * 4 blocks of 64 threads (2 waves). Each wave owns one head.
__global__ __launch_bounds__(64) void k_qkv_attn(
    const bf16* __restrict__ xn, const bf16* __restrict__ wq,
    const bf16* __restrict__ wk, const bf16* __restrict__ wv,
    const float* __restrict__ qb, const float* __restrict__ kb2,
    const float* __restrict__ vb, bf16* __restrict__ attn_out) {
  __shared__ __align__(16) bf16 s_xnt[64 * 256];   // [s][c]   32 KB
  __shared__ __align__(16) bf16 s_qt[2][64 * 32];  // [s][d]   8 KB
  __shared__ __align__(16) bf16 s_kt[2][64 * 32];  // [s][d]   8 KB
  __shared__ __align__(16) bf16 s_v [2][32 * 64];  // [d][s]   8 KB
  __shared__ __align__(16) bf16 s_p [2][16 * 64];  // [qs][ks] 4 KB

  const int tid  = threadIdx.x;
  const int lane = tid & 31;
  const int wave = tid >> 5;
  const int l15  = lane & 15;
  const int hi8  = (lane >> 4) << 3;

  const int pair = blockIdx.x & 3;
  const int win  = (blockIdx.x >> 2) & (WIN - 1);
  const int b    = blockIdx.x >> 12;
  const int h0   = (win >> 5) << 3;
  const int w0   = (win & 31) << 3;
  const int head = pair * 2 + wave;

  // stage xn tile [256 c][64 s] transposed into LDS
#pragma unroll
  for (int cc = 0; cc < 4; ++cc) {
    const int c = tid + cc * 64;
    const bf16* src = xn + (((size_t)(b * CCH + c)) << 16);
#pragma unroll
    for (int r = 0; r < 8; ++r) {
      v8bf row = *(const v8bf*)(src + (h0 + r) * 256 + w0);
#pragma unroll
      for (int j = 0; j < 8; ++j) s_xnt[(r * 8 + j) * 256 + c] = row[j];
    }
  }
  __syncthreads();

  // Q/K/V = W[32x256] * Xn[256x64] for this wave's head
#pragma unroll
  for (int pj = 0; pj < 3; ++pj) {
    const bf16*  wm = (pj == 0) ? wq : (pj == 1) ? wk : wv;
    const float* bs = (pj == 0) ? qb : (pj == 1) ? kb2 : vb;
#pragma unroll
    for (int mt = 0; mt < 2; ++mt) {
      const int m0 = mt * 16;
#pragma unroll
      for (int nt = 0; nt < 4; ++nt) {
        const int n0 = nt * 16;
        v8f acc = zero8();
#pragma unroll
        for (int kt = 0; kt < 8; ++kt) {
          v16bf a   = frag_a(wm + (size_t)(head * 32) * 256, 256, m0 + l15, kt * 32, lane);
          v16bf bfr = frag_b(s_xnt, 256, n0 + l15, kt * 32, lane);
          acc = WMMA_BF16(a, bfr, acc);
        }
        const int n = n0 + l15;
#pragma unroll
        for (int j = 0; j < 8; ++j) {
          const int m = m0 + j + hi8;
          bf16 val = (bf16)(acc[j] + bs[head * 32 + m]);
          if (pj == 0)      s_qt[wave][n * 32 + m] = val;  // transposed for A frags
          else if (pj == 1) s_kt[wave][n * 32 + m] = val;  // transposed for B frags
          else              s_v [wave][m * 64 + n] = val;  // direct for PV B frags
        }
      }
    }
  }
  __syncthreads();

  const float scale = 0.17677669529663687f;  // 1/sqrt(32)
#pragma unroll
  for (int mt = 0; mt < 4; ++mt) {
    const int m0 = mt * 16;
    v16bf aq = frag_a(s_qt[wave], 32, m0 + l15, 0, lane);
    v8f s[4];
#pragma unroll
    for (int nt = 0; nt < 4; ++nt) {
      v16bf bk = frag_b(s_kt[wave], 32, nt * 16 + l15, 0, lane);
      v8f acc = zero8();
      acc = WMMA_BF16(aq, bk, acc);
#pragma unroll
      for (int j = 0; j < 8; ++j) s[nt][j] = acc[j] * scale;
    }
    // row softmax: row M=j+hi8 lives across a 16-lane half in element j of 4 tiles
    v8f mx = s[0];
#pragma unroll
    for (int nt = 1; nt < 4; ++nt)
#pragma unroll
      for (int j = 0; j < 8; ++j) mx[j] = fmaxf(mx[j], s[nt][j]);
#pragma unroll
    for (int msk = 1; msk < 16; msk <<= 1)
#pragma unroll
      for (int j = 0; j < 8; ++j) mx[j] = fmaxf(mx[j], __shfl_xor(mx[j], msk, 32));
    v8f sum = zero8();
#pragma unroll
    for (int nt = 0; nt < 4; ++nt)
#pragma unroll
      for (int j = 0; j < 8; ++j) {
        float p = expf(s[nt][j] - mx[j]);
        s[nt][j] = p; sum[j] += p;
      }
#pragma unroll
    for (int msk = 1; msk < 16; msk <<= 1)
#pragma unroll
      for (int j = 0; j < 8; ++j) sum[j] += __shfl_xor(sum[j], msk, 32);
#pragma unroll
    for (int nt = 0; nt < 4; ++nt)
#pragma unroll
      for (int j = 0; j < 8; ++j)
        s_p[wave][(j + hi8) * 64 + nt * 16 + l15] = (bf16)(s[nt][j] / sum[j]);

    // out[16 qs][32 d] = P[16x64] * V^T[64x32]
#pragma unroll
    for (int nt = 0; nt < 2; ++nt) {
      v8f acc = zero8();
#pragma unroll
      for (int kt = 0; kt < 2; ++kt) {
        v16bf ap = frag_a(s_p[wave], 64, l15, kt * 32, lane);
        v16bf bv = frag_b(s_v[wave], 64, nt * 16 + l15, kt * 32, lane);
        acc = WMMA_BF16(ap, bv, acc);
      }
      const int chan = head * 32 + nt * 16 + l15;
#pragma unroll
      for (int j = 0; j < 8; ++j) {
        const int sq = m0 + j + hi8;
        const size_t addr = (((size_t)(b * CCH + chan)) << 16)
                          + (size_t)(h0 + (sq >> 3)) * 256 + (size_t)(w0 + (sq & 7));
        attn_out[addr] = (bf16)acc[j];
      }
    }
  }
}

// ---------------- output projection + bias + residual + scale ----------------
__global__ __launch_bounds__(256) void k_oproj(
    const bf16* __restrict__ attn, const bf16* __restrict__ wo,
    const float* __restrict__ ob, const float* __restrict__ qin,
    float* __restrict__ out) {
  __shared__ __align__(16) bf16 s_at[64 * 256];  // [s][c] 32 KB
  const int tid = threadIdx.x, lane = tid & 31, wave = tid >> 5;
  const int l15 = lane & 15, hi8 = (lane >> 4) << 3;
  const int b  = blockIdx.x >> 10;
  const int s0 = (blockIdx.x & 1023) << 6;
  {
    const int c = tid;
    const bf16* src = attn + (((size_t)(b * CCH + c)) << 16) + s0;
#pragma unroll
    for (int r = 0; r < 8; ++r) {
      v8bf row = *(const v8bf*)(src + r * 8);
#pragma unroll
      for (int j = 0; j < 8; ++j) s_at[(r * 8 + j) * 256 + c] = row[j];
    }
  }
  __syncthreads();
#pragma unroll
  for (int mt = 0; mt < 2; ++mt) {
    const int m0 = wave * 32 + mt * 16;
#pragma unroll
    for (int nt = 0; nt < 4; ++nt) {
      const int n0 = nt * 16;
      v8f acc = zero8();
#pragma unroll
      for (int kt = 0; kt < 8; ++kt) {
        v16bf a  = frag_a(wo, 256, m0 + l15, kt * 32, lane);
        v16bf bb = frag_b(s_at, 256, n0 + l15, kt * 32, lane);
        acc = WMMA_BF16(a, bb, acc);
      }
#pragma unroll
      for (int j = 0; j < 8; ++j) {
        const int o = m0 + j + hi8;
        const size_t idx = (((size_t)(b * CCH + o)) << 16) + (size_t)(s0 + n0 + l15);
        out[idx] = (acc[j] + ob[o] + qin[idx]) * 0.70710678118654752f;
      }
    }
  }
}

extern "C" void kernel_launch(void* const* d_in, const int* in_sizes, int n_in,
                              void* d_out, int out_size, void* d_ws, size_t ws_size,
                              hipStream_t stream) {
  const float* x    = (const float*)d_in[0];
  const float* gn_w = (const float*)d_in[1];
  const float* gn_b = (const float*)d_in[2];
  const float* q_w  = (const float*)d_in[3];
  const float* q_b  = (const float*)d_in[4];
  const float* k_w  = (const float*)d_in[5];
  const float* k_b  = (const float*)d_in[6];
  const float* v_w  = (const float*)d_in[7];
  const float* v_b  = (const float*)d_in[8];
  const float* o_w  = (const float*)d_in[9];
  const float* o_b  = (const float*)d_in[10];

  char* ws = (char*)d_ws;
  const size_t XN_BYTES = (size_t)BB * CCH * HW * sizeof(bf16);  // 128 MB
  const size_t AT_BYTES = XN_BYTES;                              // 128 MB
  const size_t WB_BYTES = 4ull * 65536 * sizeof(bf16);           // 512 KB
  bf16*  xn_bf   = (bf16*)ws;
  bf16*  attn_bf = (bf16*)(ws + XN_BYTES);
  bf16*  w_bf    = (bf16*)(ws + XN_BYTES + AT_BYTES);
  float* stats   = (float*)(ws + XN_BYTES + AT_BYTES + WB_BYTES);

  k_gn_stats<<<BB * GG, 256, 0, stream>>>(x, stats);
  k_norm<<<(int)(((size_t)BB * CCH * HW) / 1024), 256, 0, stream>>>(x, gn_w, gn_b, stats, xn_bf);
  k_wconv<<<(4 * 65536) / 256, 256, 0, stream>>>(q_w, k_w, v_w, o_w, w_bf);
  k_qkv_attn<<<BB * WIN * 4, 64, 0, stream>>>(xn_bf, w_bf, w_bf + 65536, w_bf + 2 * 65536,
                                              q_b, k_b, v_b, attn_bf);
  k_oproj<<<BB * (HW / 64), 256, 0, stream>>>(attn_bf, w_bf + 3 * 65536, o_b, x, (float*)d_out);
}
